// AOLLinear_51582557225235
// MI455X (gfx1250) — matrix operand order repaired and checked by reading, not compiled, not run on hardware
//
#include <hip/hip_runtime.h>

#define IN_F   2048
#define OUT_F  2048
#define BATCH  32768

typedef __attribute__((ext_vector_type(16))) _Float16 v16h;
typedef __attribute__((ext_vector_type(8)))  _Float16 v8h;
typedef __attribute__((ext_vector_type(4)))  _Float16 v4h;
typedef __attribute__((ext_vector_type(8)))  float    v8f;
typedef __attribute__((ext_vector_type(4)))  float    v4f;
typedef __attribute__((ext_vector_type(4)))  int      v4i;

#define LDS_STRIDE 56   // halves; 112B row pitch: 16B aligned, conflict-free for b128

#if defined(__has_builtin)
#if __has_builtin(__builtin_amdgcn_global_load_async_to_lds_b128)
#define HAVE_ASYNC_LDS 1
#endif
#endif
#ifndef HAVE_ASYNC_LDS
#define HAVE_ASYNC_LDS 0
#endif

// 16-byte global -> LDS copy; async DMA path on gfx1250 when available.
__device__ __forceinline__ void copy_b128_to_lds(const _Float16* g, _Float16* l) {
#if HAVE_ASYNC_LDS
    __builtin_amdgcn_global_load_async_to_lds_b128(
        (__attribute__((address_space(1))) v4i*)(unsigned long long)g,
        (__attribute__((address_space(3))) v4i*)(unsigned)(unsigned long long)l,
        0, 0);
#else
    *(v8h*)l = *(const v8h*)g;
#endif
}

__device__ __forceinline__ void wait_async_lds() {
#if HAVE_ASYNC_LDS
    asm volatile("s_wait_asynccnt 0x0" ::: "memory");
#endif
}

// ---------------------------------------------------------------------------
// 1) PTh[i][o] = (f16) P[o][i]   (transpose + convert)
// ---------------------------------------------------------------------------
__global__ void __launch_bounds__(256)
k_transpose_f16(const float* __restrict__ P, _Float16* __restrict__ PTh) {
    __shared__ float tile[32][33];
    const int i0 = blockIdx.x * 32;
    const int o0 = blockIdx.y * 32;
    const int tx = threadIdx.x & 31;
    const int ty = threadIdx.x >> 5;      // 0..7
#pragma unroll
    for (int j = 0; j < 4; ++j) {
        const int o = o0 + ty + 8 * j;
        tile[ty + 8 * j][tx] = P[(size_t)o * IN_F + i0 + tx];
    }
    __syncthreads();
#pragma unroll
    for (int j = 0; j < 4; ++j) {
        const int i = i0 + ty + 8 * j;
        PTh[(size_t)i * OUT_F + o0 + tx] = (_Float16)tile[tx][ty + 8 * j];
    }
}

// ---------------------------------------------------------------------------
// 2) Gram GEMM: C = PTh * PTh^T (contract over OUT_F), emit per-(i-tile)
//    column partial sums of |C| -> partial[i_tile][j]   (deterministic)
//    Double-buffered LDS, async global->LDS staging.
// ---------------------------------------------------------------------------
__global__ void __launch_bounds__(256)
k_gram_colsum(const _Float16* __restrict__ PT, float* __restrict__ partial) {
    __shared__ _Float16 As[2][128][LDS_STRIDE];
    __shared__ _Float16 Bs[2][128][LDS_STRIDE];
    __shared__ float    csum[2][128];

    const int j0 = blockIdx.x * 128;
    const int i0 = blockIdx.y * 128;
    const int tid  = threadIdx.x;
    const int lane = tid & 31;
    const int wid  = tid >> 5;
    const int waveM = wid & 1;            // 2 wave rows  -> 64 M each
    const int waveN = wid >> 1;           // 4 wave cols  -> 32 N each
    const int m_base = waveM * 64;
    const int n_base = waveN * 32;
    const int lrow = lane & 15;
    const bool hiL = (lane >= 16);
    const int a_k0 = hiL ? 8  : 0;
    const int a_k1 = hiL ? 24 : 16;
    const int b_k0 = hiL ? 8  : 0;

    // per-thread staging coordinates (2 x b128 per tile side)
    const int srow0 = tid >> 2;                 // 0..63
    const int srow1 = (tid + 256) >> 2;         // 64..127
    const int sc8   = (tid & 3) * 8;

    v8f acc[4][2] = {};

    // prologue: stage k=0 into buffer 0
    copy_b128_to_lds(&PT[(size_t)(i0 + srow0) * OUT_F + sc8], &As[0][srow0][sc8]);
    copy_b128_to_lds(&PT[(size_t)(i0 + srow1) * OUT_F + sc8], &As[0][srow1][sc8]);
    copy_b128_to_lds(&PT[(size_t)(j0 + srow0) * OUT_F + sc8], &Bs[0][srow0][sc8]);
    copy_b128_to_lds(&PT[(size_t)(j0 + srow1) * OUT_F + sc8], &Bs[0][srow1][sc8]);

    const int NT = OUT_F / 32;
    for (int kt = 0; kt < NT; ++kt) {
        const int cur = kt & 1;
        wait_async_lds();
        __syncthreads();

        if (kt + 1 < NT) {                      // stage next tile into other buffer
            const int kn = (kt + 1) * 32;
            const int nxt = cur ^ 1;
            copy_b128_to_lds(&PT[(size_t)(i0 + srow0) * OUT_F + kn + sc8], &As[nxt][srow0][sc8]);
            copy_b128_to_lds(&PT[(size_t)(i0 + srow1) * OUT_F + kn + sc8], &As[nxt][srow1][sc8]);
            copy_b128_to_lds(&PT[(size_t)(j0 + srow0) * OUT_F + kn + sc8], &Bs[nxt][srow0][sc8]);
            copy_b128_to_lds(&PT[(size_t)(j0 + srow1) * OUT_F + kn + sc8], &Bs[nxt][srow1][sc8]);
        }

        v16h a[4], b[2];
#pragma unroll
        for (int ma = 0; ma < 4; ++ma) {
            const _Float16* rp = &As[cur][m_base + ma * 16 + lrow][0];
            v8h lo = *(const v8h*)(rp + a_k0);
            v8h hi = *(const v8h*)(rp + a_k1);
            a[ma] = __builtin_shufflevector(lo, hi, 0,1,2,3,4,5,6,7,8,9,10,11,12,13,14,15);
        }
#pragma unroll
        for (int nb = 0; nb < 2; ++nb) {
            const _Float16* rp = &Bs[cur][n_base + nb * 16 + lrow][0];
            v8h lo = *(const v8h*)(rp + b_k0);
            v8h hi = *(const v8h*)(rp + b_k0 + 16);
            b[nb] = __builtin_shufflevector(lo, hi, 0,1,2,3,4,5,6,7,8,9,10,11,12,13,14,15);
        }
#pragma unroll
        for (int ma = 0; ma < 4; ++ma)
#pragma unroll
            for (int nb = 0; nb < 2; ++nb)
                acc[ma][nb] = __builtin_amdgcn_wmma_f32_16x16x32_f16(
                    false, a[ma], false, b[nb], (short)0, acc[ma][nb], false, false);
    }

    // reduce |acc| over the 128 i-rows of this block, per output column j
#pragma unroll
    for (int nb = 0; nb < 2; ++nb) {
        float s = 0.f;
#pragma unroll
        for (int ma = 0; ma < 4; ++ma)
#pragma unroll
            for (int r = 0; r < 8; ++r)
                s += fabsf(acc[ma][nb][r]);
        s += __shfl_down(s, 16, 32);               // combine hi/lo half-lanes
        if (!hiL) csum[waveM][n_base + nb * 16 + lrow] = s;
    }
    __syncthreads();
    if (tid < 128) {
        const float s = csum[0][tid] + csum[1][tid];
        partial[(size_t)blockIdx.y * IN_F + j0 + tid] = s;
    }
}

// ---------------------------------------------------------------------------
// 3) d[j] = rsqrt( sum_tiles partial + eps )
// ---------------------------------------------------------------------------
__global__ void k_compute_d(const float* __restrict__ partial, float* __restrict__ d) {
    const int j = blockIdx.x * blockDim.x + threadIdx.x;
    if (j < IN_F) {
        float s = 0.f;
#pragma unroll
        for (int t = 0; t < IN_F / 128; ++t)
            s += partial[(size_t)t * IN_F + j];
        d[j] = rsqrtf(s + 1e-10f);
    }
}

// ---------------------------------------------------------------------------
// 4) Wh[o][k] = (f16)( P[o][k] * d[k] )
// ---------------------------------------------------------------------------
__global__ void k_scale_w(const float* __restrict__ P, const float* __restrict__ d,
                          _Float16* __restrict__ Wh) {
    const size_t idx = (size_t)blockIdx.x * blockDim.x + threadIdx.x;
    const int kcol = (int)(idx & (IN_F - 1));
    Wh[idx] = (_Float16)(P[idx] * d[kcol]);
}

// ---------------------------------------------------------------------------
// 5) out = x @ Wh^T + bias.
//    x: f32 -> regs (prefetched one step ahead) -> f16 -> LDS.
//    Wh: async global->LDS DMA.  Double-buffered LDS pipeline.
// ---------------------------------------------------------------------------
__global__ void __launch_bounds__(256)
k_gemm_out(const float* __restrict__ X, const _Float16* __restrict__ Wh,
           const float* __restrict__ bias, float* __restrict__ out) {
    __shared__ _Float16 Xs[2][128][LDS_STRIDE];
    __shared__ _Float16 Ws[2][128][LDS_STRIDE];

    const int o0 = blockIdx.x * 128;      // N tiles fastest -> L2 reuse of x panel
    const int b0 = blockIdx.y * 128;
    const int tid  = threadIdx.x;
    const int lane = tid & 31;
    const int wid  = tid >> 5;
    const int waveM = wid & 1;
    const int waveN = wid >> 1;
    const int m_base = waveM * 64;
    const int n_base = waveN * 32;
    const int lrow = lane & 15;
    const bool hiL = (lane >= 16);
    const int a_k0 = hiL ? 8  : 0;
    const int a_k1 = hiL ? 24 : 16;
    const int b_k0 = hiL ? 8  : 0;

    // staging coordinates
    const int wrow0 = tid >> 2;                 // W: 2 x b128 / thread
    const int wrow1 = (tid + 256) >> 2;
    const int wc8   = (tid & 3) * 8;

    v8f acc[4][2] = {};
    v4f xr[4];

    // prologue: W tile k=0 -> async DMA; X tile k=0 -> regs
    copy_b128_to_lds(&Wh[(size_t)(o0 + wrow0) * IN_F + wc8], &Ws[0][wrow0][wc8]);
    copy_b128_to_lds(&Wh[(size_t)(o0 + wrow1) * IN_F + wc8], &Ws[0][wrow1][wc8]);
#pragma unroll
    for (int t = 0; t < 4; ++t) {
        const int v   = tid + 256 * t;          // 0..1023
        const int row = v >> 3;
        const int c4  = (v & 7) * 4;
        xr[t] = *(const v4f*)&X[(size_t)(b0 + row) * IN_F + c4];
    }

    const int NT = IN_F / 32;
    for (int kt = 0; kt < NT; ++kt) {
        const int cur = kt & 1;

        // publish X regs -> LDS (current buffer)
#pragma unroll
        for (int t = 0; t < 4; ++t) {
            const int v   = tid + 256 * t;
            const int row = v >> 3;
            const int c4  = (v & 7) * 4;
            v4h h;
            h[0] = (_Float16)xr[t][0]; h[1] = (_Float16)xr[t][1];
            h[2] = (_Float16)xr[t][2]; h[3] = (_Float16)xr[t][3];
            *(v4h*)&Xs[cur][row][c4] = h;
        }
        wait_async_lds();
        __syncthreads();

        if (kt + 1 < NT) {                      // stage next step
            const int kn  = (kt + 1) * 32;
            const int nxt = cur ^ 1;
            copy_b128_to_lds(&Wh[(size_t)(o0 + wrow0) * IN_F + kn + wc8], &Ws[nxt][wrow0][wc8]);
            copy_b128_to_lds(&Wh[(size_t)(o0 + wrow1) * IN_F + kn + wc8], &Ws[nxt][wrow1][wc8]);
#pragma unroll
            for (int t = 0; t < 4; ++t) {
                const int v   = tid + 256 * t;
                const int row = v >> 3;
                const int c4  = (v & 7) * 4;
                xr[t] = *(const v4f*)&X[(size_t)(b0 + row) * IN_F + kn + c4];
            }
        }

        v16h a[4], b[2];
#pragma unroll
        for (int ma = 0; ma < 4; ++ma) {
            const _Float16* rp = &Xs[cur][m_base + ma * 16 + lrow][0];
            v8h lo = *(const v8h*)(rp + a_k0);
            v8h hi = *(const v8h*)(rp + a_k1);
            a[ma] = __builtin_shufflevector(lo, hi, 0,1,2,3,4,5,6,7,8,9,10,11,12,13,14,15);
        }
#pragma unroll
        for (int nb = 0; nb < 2; ++nb) {
            const _Float16* rp = &Ws[cur][n_base + nb * 16 + lrow][0];
            v8h lo = *(const v8h*)(rp + b_k0);
            v8h hi = *(const v8h*)(rp + b_k0 + 16);
            b[nb] = __builtin_shufflevector(lo, hi, 0,1,2,3,4,5,6,7,8,9,10,11,12,13,14,15);
        }
#pragma unroll
        for (int ma = 0; ma < 4; ++ma)
#pragma unroll
            for (int nb = 0; nb < 2; ++nb)
                acc[ma][nb] = __builtin_amdgcn_wmma_f32_16x16x32_f16(
                    false, a[ma], false, b[nb], (short)0, acc[ma][nb], false, false);
    }

    // epilogue: C layout -> lane holds col (lrow), rows r + (hiL?8:0)
#pragma unroll
    for (int nb = 0; nb < 2; ++nb) {
        const int col = o0 + n_base + nb * 16 + lrow;
        const float bv = bias[col];
#pragma unroll
        for (int ma = 0; ma < 4; ++ma) {
            const int rbase = b0 + m_base + ma * 16 + (hiL ? 8 : 0);
#pragma unroll
            for (int r = 0; r < 8; ++r)
                out[(size_t)(rbase + r) * OUT_F + col] = acc[ma][nb][r] + bv;
        }
    }
}

// ---------------------------------------------------------------------------
extern "C" void kernel_launch(void* const* d_in, const int* in_sizes, int n_in,
                              void* d_out, int out_size, void* d_ws, size_t ws_size,
                              hipStream_t stream) {
    const float* x    = (const float*)d_in[0];
    const float* P    = (const float*)d_in[1];
    const float* bias = (const float*)d_in[2];
    float* out = (float*)d_out;

    char* ws = (char*)d_ws;
    float*    partial = (float*)ws;                              // 16*2048*4   = 128 KB
    float*    dvec    = (float*)(ws + 131072);                   // 8 KB
    _Float16* PTh     = (_Float16*)(ws + 139264);                // 8 MB
    _Float16* Wh      = (_Float16*)(ws + 139264 + 8u * 1024 * 1024); // 8 MB

    k_transpose_f16<<<dim3(IN_F / 32, OUT_F / 32), 256, 0, stream>>>(P, PTh);
    k_gram_colsum<<<dim3(IN_F / 128, IN_F / 128), 256, 0, stream>>>(PTh, partial);
    k_compute_d<<<IN_F / 256, 256, 0, stream>>>(partial, dvec);
    k_scale_w<<<(OUT_F * IN_F) / 256, 256, 0, stream>>>(P, dvec, Wh);
    k_gemm_out<<<dim3(OUT_F / 128, BATCH / 128), 256, 0, stream>>>(x, Wh, bias, out);
}
